// InfoQuantizer_8048768713193
// MI455X (gfx1250) — matrix operand order repaired and checked by reading, not compiled, not run on hardware
//
#include <hip/hip_runtime.h>
#include <math.h>

// ---------------------------------------------------------------------------
// InfoQuantizer forward for MI455X (gfx1250, wave32, WMMA).
// All GEMMs run on v_wmma_f32_16x16x32_bf16 with f32 accumulation.
// B-matrices (weights, log(E)^T) are pre-packed into WMMA fragment-major
// layout so the hot loop issues only coalesced 128-bit global loads with
// immediate offsets. Encoder layers use a 32-row block tile (each B fragment
// feeds 2 WMMAs), a double-buffered LDS A-stage (1 barrier per K-step), and
// an explicit 2-deep B-fragment register pipeline (loads overlap WMMAs).
// ---------------------------------------------------------------------------

typedef __attribute__((ext_vector_type(16))) __bf16 v16bf;
typedef __attribute__((ext_vector_type(8)))  float  v8f;

union Frag {
  v16bf v;
  unsigned int w[8];
};

__device__ __forceinline__ unsigned short f2bf(float f) {
  union { float f; unsigned int u; } c; c.f = f;
  unsigned int u = c.u;
  u += 0x7FFFu + ((u >> 16) & 1u);   // round-to-nearest-even
  return (unsigned short)(u >> 16);
}

__device__ __forceinline__ void load_frag(Frag& fr, const unsigned short* p) {
  const uint4* bp = (const uint4*)p;
  *(uint4*)&fr.w[0] = bp[0];
  *(uint4*)&fr.w[4] = bp[1];
}

// ---------------------------------------------------------------------------
// Packing / conversion kernels (run once per launch; bandwidth-trivial).
// ---------------------------------------------------------------------------

__global__ void cvt_bf16_kernel(const float* __restrict__ src,
                                unsigned short* __restrict__ dst, int n) {
  int i = blockIdx.x * 256 + threadIdx.x;
  if (i < n) dst[i] = f2bf(src[i]);
}

__global__ void logf_kernel(const float* __restrict__ src,
                            float* __restrict__ dst, int n) {
  int i = blockIdx.x * 256 + threadIdx.x;
  if (i < n) dst[i] = logf(src[i]);
}

// Pack row-major W[K][N] into WMMA B-fragment order:
// dst[((ks*ntiles + nt)*32 + lane)*16 + e] = bf16(W[k][n])
//   k = ks*32 + (e>=8 ? 16 : 0) + (lane>=16 ? 8 : 0) + (e&7),  n = nt*16 + (lane&15)
__global__ void pack_b_kernel(const float* __restrict__ W,
                              unsigned short* __restrict__ dst, int K, int N) {
  int gid = blockIdx.x * 256 + threadIdx.x;
  if (gid >= K * N) return;
  int f = gid >> 9;          // fragment index (512 elems each)
  int r = gid & 511;
  int lane = r >> 4;
  int e = r & 15;
  int ntiles = N >> 4;
  int ks = f / ntiles;
  int nt = f - ks * ntiles;
  int k = ks * 32 + ((e >> 3) << 4) + ((lane >> 4) << 3) + (e & 7);
  int n = (nt << 4) + (lane & 15);
  dst[gid] = f2bf(W[(size_t)k * N + n]);
}

// Same, but source is S[N][K] (i.e. B[k][n] = S[n][k]) -> packs log(E)^T.
__global__ void pack_bT_kernel(const float* __restrict__ S,
                               unsigned short* __restrict__ dst, int K, int N) {
  int gid = blockIdx.x * 256 + threadIdx.x;
  if (gid >= K * N) return;
  int f = gid >> 9;
  int r = gid & 511;
  int lane = r >> 4;
  int e = r & 15;
  int ntiles = N >> 4;
  int ks = f / ntiles;
  int nt = f - ks * ntiles;
  int k = ks * 32 + ((e >> 3) << 4) + ((lane >> 4) << 3) + (e & 7);
  int n = (nt << 4) + (lane & 15);
  dst[gid] = f2bf(S[(size_t)n * K + k]);
}

// ---------------------------------------------------------------------------
// Fused GEMM (bf16 WMMA) + LayerNorm + ReLU.
// Block: 256 threads = 8 waves. Block tile: 32 rows x 1024 cols.
// ---------------------------------------------------------------------------
__global__ __launch_bounds__(256)
void gemm_ln_relu_kernel(const unsigned short* __restrict__ A,   // bf16 [N][K]
                         const unsigned short* __restrict__ pckB,
                         const float* __restrict__ gam,
                         const float* __restrict__ bet,
                         unsigned short* __restrict__ out,       // bf16 [N][1024]
                         int K) {
  const int C = 1024;
  __shared__ unsigned int ldsA[2][512];       // 2 x (32 rows x 32 bf16)
  __shared__ float partS[256], partS2[256];   // [wave][row(32)]
  const int tid  = threadIdx.x;
  const int wave = tid >> 5, lane = tid & 31;
  const int hf   = lane >> 4, ln = lane & 15;
  const int rowbase = blockIdx.x * 32;
  const int colbase = wave * 128;
  const int ksteps  = K >> 5;

  v8f vz = {0.f, 0.f, 0.f, 0.f, 0.f, 0.f, 0.f, 0.f};
  v8f acc[2][8];
#pragma unroll
  for (int m = 0; m < 2; ++m)
#pragma unroll
    for (int t = 0; t < 8; ++t) acc[m][t] = vz;

  // per-thread A staging pointers (advance by 32 elems = 64B per K-step)
  const unsigned short* aP0 = A + (size_t)(rowbase + (tid >> 4)) * K + (tid & 15) * 2;
  const unsigned short* aP1 = A + (size_t)(rowbase + 16 + (tid >> 4)) * K + (tid & 15) * 2;
  // per-wave B base pointer (advance by 64 tiles * 512 elems per K-step)
  const unsigned short* bbase = pckB + ((size_t)(wave * 8) * 32 + lane) * 16;

  // prologue: stage K-step 0
  ldsA[0][tid]       = *(const unsigned int*)aP0;
  ldsA[0][tid + 256] = *(const unsigned int*)aP1;
  aP0 += 32; aP1 += 32;
  __syncthreads();

  for (int ks = 0; ks < ksteps; ++ks) {
    const int p = ks & 1;
    // A fragments for both 16-row halves
    Frag af0, af1;
    const int b0 = ln * 16 + hf * 4;
    const int b1 = (16 + ln) * 16 + hf * 4;
#pragma unroll
    for (int j = 0; j < 4; ++j) {
      af0.w[j]     = ldsA[p][b0 + j];
      af0.w[4 + j] = ldsA[p][b0 + 8 + j];
      af1.w[j]     = ldsA[p][b1 + j];
      af1.w[4 + j] = ldsA[p][b1 + 8 + j];
    }
    // prefetch next A tile into the other buffer (overlaps the WMMAs)
    if (ks + 1 < ksteps) {
      ldsA[p ^ 1][tid]       = *(const unsigned int*)aP0;
      ldsA[p ^ 1][tid + 256] = *(const unsigned int*)aP1;
      aP0 += 32; aP1 += 32;
    }
    // 8 n-tiles, 2-deep software pipeline on B fragments
    const unsigned short* bptr = bbase;
    bbase += 64 * 512;                      // next K-step
    Frag bf0, bf1;
    load_frag(bf0, bptr);
#pragma unroll
    for (int t = 0; t < 8; t += 2) {
      load_frag(bf1, bptr + (t + 1) * 512);
      acc[0][t] = __builtin_amdgcn_wmma_f32_16x16x32_bf16(
          false, af0.v, false, bf0.v, (short)0, acc[0][t], false, false);
      acc[1][t] = __builtin_amdgcn_wmma_f32_16x16x32_bf16(
          false, af1.v, false, bf0.v, (short)0, acc[1][t], false, false);
      if (t + 2 < 8) load_frag(bf0, bptr + (t + 2) * 512);
      acc[0][t + 1] = __builtin_amdgcn_wmma_f32_16x16x32_bf16(
          false, af0.v, false, bf1.v, (short)0, acc[0][t + 1], false, false);
      acc[1][t + 1] = __builtin_amdgcn_wmma_f32_16x16x32_bf16(
          false, af1.v, false, bf1.v, (short)0, acc[1][t + 1], false, false);
    }
    __syncthreads();
  }

  // LayerNorm over C=1024 (spread over 8 waves) + ReLU, output bf16.
#pragma unroll
  for (int m = 0; m < 2; ++m) {
#pragma unroll
    for (int v = 0; v < 8; ++v) {
      float s = 0.f, s2 = 0.f;
#pragma unroll
      for (int t = 0; t < 8; ++t) { float x = acc[m][t][v]; s += x; s2 += x * x; }
#pragma unroll
      for (int off = 1; off < 16; off <<= 1) {
        s  += __shfl_xor(s,  off, 32);
        s2 += __shfl_xor(s2, off, 32);
      }
      if (ln == 0) {
        int row = m * 16 + v + hf * 8;
        partS [wave * 32 + row] = s;
        partS2[wave * 32 + row] = s2;
      }
    }
  }
  __syncthreads();
  const float invC = 1.0f / (float)C;
#pragma unroll
  for (int m = 0; m < 2; ++m) {
#pragma unroll
    for (int v = 0; v < 8; ++v) {
      int row = m * 16 + v + hf * 8;
      float S = 0.f, S2 = 0.f;
#pragma unroll
      for (int w = 0; w < 8; ++w) { S += partS[w * 32 + row]; S2 += partS2[w * 32 + row]; }
      float mu   = S * invC;
      float var  = S2 * invC - mu * mu;
      float rstd = rsqrtf(var + 1e-5f);
#pragma unroll
      for (int t = 0; t < 8; ++t) {
        int col = colbase + t * 16 + ln;
        float x = (acc[m][t][v] - mu) * rstd * gam[col] + bet[col];
        x = x > 0.f ? x : 0.f;
        out[(size_t)(rowbase + row) * C + col] = f2bf(x);
      }
    }
  }
}

// ---------------------------------------------------------------------------
// Head: z = h@W5 + b5 (K=1024, D=256), then log_softmax -> p, ep=exp(p),
// self = sum(ep*p). Wave w owns 32 cols (2 WMMA tiles).
// ---------------------------------------------------------------------------
__global__ __launch_bounds__(256)
void head_kernel(const unsigned short* __restrict__ A,    // bf16 [N][1024]
                 const unsigned short* __restrict__ pckB, // W5 packed
                 const float* __restrict__ bias,
                 float* __restrict__ zout,
                 float* __restrict__ pout,
                 unsigned short* __restrict__ epout,
                 float* __restrict__ selfout) {
  const int K = 1024, D = 256;
  __shared__ unsigned int ldsA[256];
  __shared__ float part[128];
  const int tid  = threadIdx.x;
  const int wave = tid >> 5, lane = tid & 31;
  const int hf   = lane >> 4, ln = lane & 15;
  const int rowbase = blockIdx.x * 16;
  const int colbase = wave * 32;

  v8f vz = {0.f, 0.f, 0.f, 0.f, 0.f, 0.f, 0.f, 0.f};
  v8f acc[2]; acc[0] = vz; acc[1] = vz;

  const unsigned short* aP = A + (size_t)(rowbase + (tid >> 4)) * K + (tid & 15) * 2;
  const unsigned short* bbase = pckB + ((size_t)(wave * 2) * 32 + lane) * 16;

  for (int ks = 0; ks < 32; ++ks) {
    ldsA[tid] = *(const unsigned int*)aP;
    aP += 32;
    __syncthreads();
    Frag af;
    const int base = ln * 16 + hf * 4;
#pragma unroll
    for (int j = 0; j < 4; ++j) { af.w[j] = ldsA[base + j]; af.w[4 + j] = ldsA[base + 8 + j]; }
    Frag bf0, bf1;
    load_frag(bf0, bbase);
    load_frag(bf1, bbase + 512);
    bbase += 16 * 512;
    acc[0] = __builtin_amdgcn_wmma_f32_16x16x32_bf16(
        false, af.v, false, bf0.v, (short)0, acc[0], false, false);
    acc[1] = __builtin_amdgcn_wmma_f32_16x16x32_bf16(
        false, af.v, false, bf1.v, (short)0, acc[1], false, false);
    __syncthreads();
  }

  // bias + z store + row max
#pragma unroll
  for (int v = 0; v < 8; ++v) {
    int row = v + hf * 8;
    float mx = -3.4e38f;
#pragma unroll
    for (int t = 0; t < 2; ++t) {
      int col = colbase + t * 16 + ln;
      float z = acc[t][v] + bias[col];
      acc[t][v] = z;
      zout[(size_t)(rowbase + row) * D + col] = z;
      mx = fmaxf(mx, z);
    }
#pragma unroll
    for (int off = 1; off < 16; off <<= 1) mx = fmaxf(mx, __shfl_xor(mx, off, 32));
    if (ln == 0) part[wave * 16 + row] = mx;
  }
  __syncthreads();
  float tmax[8];
#pragma unroll
  for (int v = 0; v < 8; ++v) {
    int row = v + hf * 8;
    float m = part[row];
#pragma unroll
    for (int w = 1; w < 8; ++w) m = fmaxf(m, part[w * 16 + row]);
    tmax[v] = m;
  }
  __syncthreads();
  // sum exp
#pragma unroll
  for (int v = 0; v < 8; ++v) {
    int row = v + hf * 8;
    float se = expf(acc[0][v] - tmax[v]) + expf(acc[1][v] - tmax[v]);
#pragma unroll
    for (int off = 1; off < 16; off <<= 1) se += __shfl_xor(se, off, 32);
    if (ln == 0) part[wave * 16 + row] = se;
  }
  __syncthreads();
  float logZ[8];
#pragma unroll
  for (int v = 0; v < 8; ++v) {
    int row = v + hf * 8;
    float S = 0.f;
#pragma unroll
    for (int w = 0; w < 8; ++w) S += part[w * 16 + row];
    logZ[v] = tmax[v] + logf(S);
  }
  __syncthreads();
  // p, ep, self-term
#pragma unroll
  for (int v = 0; v < 8; ++v) {
    int row = v + hf * 8;
    float st = 0.f;
#pragma unroll
    for (int t = 0; t < 2; ++t) {
      int col = colbase + t * 16 + ln;
      size_t o = (size_t)(rowbase + row) * D + col;
      float p = acc[t][v] - logZ[v];
      pout[o] = p;
      float ep = expf(p);
      epout[o] = f2bf(ep);
      st += ep * p;
    }
#pragma unroll
    for (int off = 1; off < 16; off <<= 1) st += __shfl_xor(st, off, 32);
    if (ln == 0) part[wave * 16 + row] = st;
  }
  __syncthreads();
  if (wave == 0 && ln == 0) {
#pragma unroll
    for (int v = 0; v < 8; ++v) {
      int row = v + hf * 8;
      float S = 0.f;
#pragma unroll
      for (int w = 0; w < 8; ++w) S += part[w * 16 + row];
      selfout[rowbase + row] = S;
    }
  }
}

// ---------------------------------------------------------------------------
// Scores = ep @ log(E)^T (K=256, Mcodes=1024) via WMMA; row argmax (== argmin
// of KL divergence, ties -> lowest index); q = E[idx]; masked KL partials.
// ---------------------------------------------------------------------------
__global__ __launch_bounds__(256)
void quant_kernel(const unsigned short* __restrict__ ep,    // bf16 [N][256]
                  const unsigned short* __restrict__ pckL,  // log(E)^T packed
                  const float* __restrict__ logE32,         // [1024][256]
                  const float* __restrict__ emb,            // [1024][256]
                  const float* __restrict__ pout,           // [N][256]
                  const float* __restrict__ selfb,          // [N]
                  const float* __restrict__ masks,          // [N]
                  float* __restrict__ qout,                 // [N][256]
                  float* __restrict__ lossPart) {           // [N/16]
  const int K = 256, D = 256;
  __shared__ unsigned int ldsA[256];
  __shared__ float partV[128];
  __shared__ int   partI[128];
  __shared__ int   finalIdx[16];
  __shared__ float partR[8];
  const int tid  = threadIdx.x;
  const int wave = tid >> 5, lane = tid & 31;
  const int hf   = lane >> 4, ln = lane & 15;
  const int rowbase = blockIdx.x * 16;
  const int colbase = wave * 128;

  v8f vz = {0.f, 0.f, 0.f, 0.f, 0.f, 0.f, 0.f, 0.f};
  v8f acc[8];
#pragma unroll
  for (int t = 0; t < 8; ++t) acc[t] = vz;

  const unsigned short* aP = ep + (size_t)(rowbase + (tid >> 4)) * K + (tid & 15) * 2;
  const unsigned short* bbase = pckL + ((size_t)(wave * 8) * 32 + lane) * 16;

  for (int ks = 0; ks < (K >> 5); ++ks) {
    ldsA[tid] = *(const unsigned int*)aP;
    aP += 32;
    __syncthreads();
    Frag af;
    const int base = ln * 16 + hf * 4;
#pragma unroll
    for (int j = 0; j < 4; ++j) { af.w[j] = ldsA[base + j]; af.w[4 + j] = ldsA[base + 8 + j]; }
    const unsigned short* bptr = bbase;
    bbase += 64 * 512;
    Frag bf0, bf1;
    load_frag(bf0, bptr);
#pragma unroll
    for (int t = 0; t < 8; t += 2) {
      load_frag(bf1, bptr + (t + 1) * 512);
      acc[t] = __builtin_amdgcn_wmma_f32_16x16x32_bf16(
          false, af.v, false, bf0.v, (short)0, acc[t], false, false);
      if (t + 2 < 8) load_frag(bf0, bptr + (t + 2) * 512);
      acc[t + 1] = __builtin_amdgcn_wmma_f32_16x16x32_bf16(
          false, af.v, false, bf1.v, (short)0, acc[t + 1], false, false);
    }
    __syncthreads();
  }

  // argmax over 1024 codes (ties -> lowest index, matching jnp.argmin)
#pragma unroll
  for (int v = 0; v < 8; ++v) {
    float bv = -3.4e38f; int bi = 0;
#pragma unroll
    for (int t = 0; t < 8; ++t) {
      float x = acc[t][v];
      int m = colbase + t * 16 + ln;
      if (x > bv || (x == bv && m < bi)) { bv = x; bi = m; }
    }
#pragma unroll
    for (int off = 1; off < 16; off <<= 1) {
      float ov = __shfl_xor(bv, off, 32);
      int   oi = __shfl_xor(bi, off, 32);
      if (ov > bv || (ov == bv && oi < bi)) { bv = ov; bi = oi; }
    }
    if (ln == 0) {
      int row = v + hf * 8;
      partV[wave * 16 + row] = bv;
      partI[wave * 16 + row] = bi;
    }
  }
  __syncthreads();
  if (tid < 16) {
    float bv = partV[tid]; int bi = partI[tid];
#pragma unroll
    for (int w = 1; w < 8; ++w) {
      float ov = partV[w * 16 + tid]; int oi = partI[w * 16 + tid];
      if (ov > bv || (ov == bv && oi < bi)) { bv = ov; bi = oi; }
    }
    finalIdx[tid] = bi;
  }
  __syncthreads();

  // q = E[idx]; kl = self - sum(ep*logE[idx]); accumulate masked loss
  float lossAcc = 0.f;
  for (int r = 0; r < 16; ++r) {
    int row = rowbase + r;
    int idx = finalIdx[r];
    int d = tid;                       // 256 threads <-> D=256
    size_t eo = (size_t)idx * D + d;
    float lq = logE32[eo];
    qout[(size_t)row * D + d] = emb[eo];
    float pv = pout[(size_t)row * D + d];
    float partial = expf(pv) * lq;
#pragma unroll
    for (int off = 1; off < 32; off <<= 1) partial += __shfl_xor(partial, off, 32);
    if (lane == 0) partR[wave] = partial;
    __syncthreads();
    if (tid == 0) {
      float dot = 0.f;
#pragma unroll
      for (int w = 0; w < 8; ++w) dot += partR[w];
      lossAcc += (selfb[row] - dot) * masks[row];
    }
    __syncthreads();
  }
  if (tid == 0) lossPart[blockIdx.x] = lossAcc;
}

// Deterministic final loss reduction (fixed order, no atomics).
__global__ __launch_bounds__(256)
void loss_reduce_kernel(const float* __restrict__ parts,
                        float* __restrict__ lossout, int n) {
  __shared__ float buf[256];
  int tid = threadIdx.x;
  float s = 0.f;
  for (int i = tid; i < n; i += 256) s += parts[i];
  buf[tid] = s;
  __syncthreads();
  if (tid == 0) {
    float t = 0.f;
    for (int i = 0; i < 256; ++i) t += buf[i];
    *lossout = t * (0.25f / 16.0f);   // 0.25 * mean over B=16, summed over T
  }
}

// ---------------------------------------------------------------------------

extern "C" void kernel_launch(void* const* d_in, const int* in_sizes, int n_in,
                              void* d_out, int out_size, void* d_ws, size_t ws_size,
                              hipStream_t stream) {
  const int N = 16 * 1024;      // B*T
  const int Cin = 512, C = 1024, D = 256, M = 1024;

  const float* x     = (const float*)d_in[0];
  const float* masks = (const float*)d_in[1];
  const float* W1 = (const float*)d_in[2];
  const float* g1 = (const float*)d_in[3];
  const float* b1 = (const float*)d_in[4];
  const float* W2 = (const float*)d_in[5];
  const float* g2 = (const float*)d_in[6];
  const float* b2 = (const float*)d_in[7];
  const float* W3 = (const float*)d_in[8];
  const float* g3 = (const float*)d_in[9];
  const float* b3 = (const float*)d_in[10];
  const float* W4 = (const float*)d_in[11];
  const float* g4 = (const float*)d_in[12];
  const float* b4 = (const float*)d_in[13];
  const float* W5 = (const float*)d_in[14];
  const float* b5 = (const float*)d_in[15];
  const float* emb = (const float*)d_in[16];

  char* base = (char*)d_ws;
  size_t off = 0;
  auto take = [&](size_t bytes) -> char* {
    char* p = base + off;
    off += (bytes + 255) & ~(size_t)255;
    return p;
  };
  unsigned short* xbf  = (unsigned short*)take((size_t)N * Cin * 2); // reused as ep
  unsigned short* h0   = (unsigned short*)take((size_t)N * C * 2);   // reused as p (f32, N*D)
  unsigned short* h1   = (unsigned short*)take((size_t)N * C * 2);
  unsigned short* pck1 = (unsigned short*)take((size_t)Cin * C * 2);
  unsigned short* pck2 = (unsigned short*)take((size_t)C * C * 2);
  unsigned short* pck3 = (unsigned short*)take((size_t)C * C * 2);
  unsigned short* pck4 = (unsigned short*)take((size_t)C * C * 2);
  unsigned short* pck5 = (unsigned short*)take((size_t)C * D * 2);
  unsigned short* pckL = (unsigned short*)take((size_t)D * M * 2);
  float* logE32   = (float*)take((size_t)M * D * 4);
  float* selfb    = (float*)take((size_t)N * 4);
  float* lossPart = (float*)take((size_t)(N / 16) * 4);

  unsigned short* epbuf = xbf;        // x consumed after layer 1
  float* pbuf = (float*)h0;           // h0 consumed after layer 4

  float* zout    = (float*)d_out;
  float* qout    = zout + (size_t)N * D;
  float* lossout = zout + (size_t)2 * N * D;

  dim3 blk(256);
  auto gs = [](int n) { return dim3((unsigned)((n + 255) / 256)); };

  // Pack phase
  cvt_bf16_kernel<<<gs(N * Cin), blk, 0, stream>>>(x, xbf, N * Cin);
  pack_b_kernel<<<gs(Cin * C), blk, 0, stream>>>(W1, pck1, Cin, C);
  pack_b_kernel<<<gs(C * C),   blk, 0, stream>>>(W2, pck2, C, C);
  pack_b_kernel<<<gs(C * C),   blk, 0, stream>>>(W3, pck3, C, C);
  pack_b_kernel<<<gs(C * C),   blk, 0, stream>>>(W4, pck4, C, C);
  pack_b_kernel<<<gs(C * D),   blk, 0, stream>>>(W5, pck5, C, D);
  logf_kernel<<<gs(M * D), blk, 0, stream>>>(emb, logE32, M * D);
  pack_bT_kernel<<<gs(D * M), blk, 0, stream>>>(logE32, pckL, D, M);

  // Encoder (32-row tiles) + head + quantize (16-row tiles)
  dim3 grid32(N / 32);
  dim3 grid16(N / 16);
  gemm_ln_relu_kernel<<<grid32, blk, 0, stream>>>(xbf, pck1, g1, b1, h0, Cin);
  gemm_ln_relu_kernel<<<grid32, blk, 0, stream>>>(h0,  pck2, g2, b2, h1, C);
  gemm_ln_relu_kernel<<<grid32, blk, 0, stream>>>(h1,  pck3, g3, b3, h0, C);
  gemm_ln_relu_kernel<<<grid32, blk, 0, stream>>>(h0,  pck4, g4, b4, h1, C);
  head_kernel<<<grid16, blk, 0, stream>>>(h1, pck5, b5, zout, pbuf, epbuf, selfb);
  quant_kernel<<<grid16, blk, 0, stream>>>(epbuf, pckL, logE32, emb, pbuf, selfb,
                                           masks, qout, lossPart);
  loss_reduce_kernel<<<dim3(1), blk, 0, stream>>>(lossPart, lossout, N / 16);
}
